// Transformer_79551384256826
// MI455X (gfx1250) — matrix operand order repaired and checked by reading, not compiled, not run on hardware
//
#include <hip/hip_runtime.h>
#include <hip/hip_bf16.h>
#include <cstdint>
#include <cstddef>

// ---------------------------------------------------------------------------
// CDNA5 (gfx1250) transformer forward, bf16 WMMA + fp32 accumulate.
// ---------------------------------------------------------------------------

typedef __bf16 bf16;
typedef __attribute__((ext_vector_type(16))) __bf16 v16bf;
typedef __attribute__((ext_vector_type(8)))  __bf16 v8bf;
typedef __attribute__((ext_vector_type(8)))  float  v8f;

#define DMODEL 1024
#define SEQ    1024
#define NHEAD  16
#define HDIM   64
#define NLAYER 6
#define BATCH  2
#define MROWS  (BATCH * SEQ)   // 2048
#define VOCAB  32000

__device__ __forceinline__ bf16 to_bf16(float f) {
  unsigned u = __float_as_uint(f);
  u += 0x7FFFu + ((u >> 16) & 1u);             // round-to-nearest-even
  unsigned short s = (unsigned short)(u >> 16);
  return __builtin_bit_cast(bf16, s);
}

struct AFragBits { v8bf lo, hi; };
__device__ __forceinline__ v16bf make_a(v8bf lo, v8bf hi) {
  AFragBits t{lo, hi};
  return __builtin_bit_cast(v16bf, t);
}

// ---------------------------------------------------------------------------
// Embedding gather: h[row][d] = tok_emb[x[row]][d] + pos_emb[row%T][d]
// Also emits bf16 copy (post-LN arch => QKV GEMMs read the raw residual).
// ---------------------------------------------------------------------------
__global__ void embed_kernel(const int* __restrict__ x,
                             const float* __restrict__ tok,
                             const float* __restrict__ pos,
                             float* __restrict__ h, bf16* __restrict__ hb) {
  int row = blockIdx.x;
  int t = row & (SEQ - 1);
  int token = x[row];
  const float* te = tok + (size_t)token * DMODEL;
  const float* pe = pos + (size_t)t * DMODEL;
  for (int d = threadIdx.x; d < DMODEL; d += blockDim.x) {
    float v = te[d] + pe[d];
    h[(size_t)row * DMODEL + d] = v;
    hb[(size_t)row * DMODEL + d] = to_bf16(v);
  }
}

// ---------------------------------------------------------------------------
// Weight pack: fp32 [K][N] row-major -> bf16 [N][K], LDS-tiled 32x32 transpose
// so both global read and write phases are coalesced. K, N multiples of 32.
// ---------------------------------------------------------------------------
__global__ void pack_wt_kernel(const float* __restrict__ W, bf16* __restrict__ Wt,
                               int K, int N) {
  __shared__ bf16 tile[32][33];
  int kt = blockIdx.x * 32;
  int nt = blockIdx.y * 32;
  int tx = threadIdx.x & 31;        // fast dim
  int ty = threadIdx.x >> 5;        // 8 rows per pass
#pragma unroll
  for (int j = 0; j < 4; ++j) {
    int k = kt + ty + j * 8;
    tile[tx][ty + j * 8] = to_bf16(W[(size_t)k * N + nt + tx]);  // coalesced read
  }
  __syncthreads();
#pragma unroll
  for (int j = 0; j < 4; ++j) {
    int n = nt + ty + j * 8;
    Wt[(size_t)n * K + kt + tx] = tile[ty + j * 8][tx];          // coalesced write
  }
}

// ---------------------------------------------------------------------------
// GEMM: C[M][N] = A[M][K](bf16) * Bt[N][K](bf16)^T + bias, optional ReLU,
// optional fp32 and/or bf16 outputs.
// Block = 4 waves; each wave owns a 32(M) x 64(N) tile: 2 A-frags x 4 B-frags
// -> 8 WMMAs per 32-wide k-step, 2.7x the operand reuse of a 16x16 tile.
// Software prefetch (global_prefetch_b8) one 128-element chunk ahead.
// ---------------------------------------------------------------------------
__global__ void gemm_bf16_wmma(const bf16* __restrict__ A,
                               const bf16* __restrict__ Bt,
                               const float* __restrict__ bias,
                               float* __restrict__ Cf, bf16* __restrict__ Cb,
                               int M, int N, int K, int relu) {
  int lane = threadIdx.x & 31;
  int wave = threadIdx.x >> 5;
  int half = lane >> 4;
  int ln   = lane & 15;
  int m0 = blockIdx.x * 32;
  int n0 = blockIdx.y * 256 + wave * 64;

  const bf16* a0 = A + (size_t)(m0 + ln) * K + 16 * half;       // A-frag rows 0..15
  const bf16* a1 = A + (size_t)(m0 + 16 + ln) * K + 16 * half;  // A-frag rows 16..31
  const bf16* b0 = Bt + (size_t)(n0 + ln) * K + 16 * half;      // B-frag cols, +j*16*K

  v8f acc[2][4] = {};
  const size_t bstride = (size_t)16 * K;

  for (int k0 = 0; k0 < K; k0 += 128) {
    // prefetch next chunk of each stream into cache (global_prefetch_b8)
    __builtin_prefetch(a0 + k0 + 128, 0, 3);
    __builtin_prefetch(a1 + k0 + 128, 0, 3);
#pragma unroll
    for (int j = 0; j < 4; ++j)
      __builtin_prefetch(b0 + (size_t)j * bstride + k0 + 128, 0, 3);

#pragma unroll
    for (int kk = 0; kk < 128; kk += 32) {
      int k = k0 + kk;
      v16bf A0 = make_a(*(const v8bf*)(a0 + k), *(const v8bf*)(a0 + k + 8));
      v16bf A1 = make_a(*(const v8bf*)(a1 + k), *(const v8bf*)(a1 + k + 8));
#pragma unroll
      for (int j = 0; j < 4; ++j) {
        v16bf B = *(const v16bf*)(b0 + (size_t)j * bstride + k);
        acc[0][j] = __builtin_amdgcn_wmma_f32_16x16x32_bf16(
            false, A0, false, B, (short)0, acc[0][j], false, false);
        acc[1][j] = __builtin_amdgcn_wmma_f32_16x16x32_bf16(
            false, A1, false, B, (short)0, acc[1][j], false, false);
      }
    }
  }

#pragma unroll
  for (int j = 0; j < 4; ++j) {
    int col = n0 + j * 16 + ln;
    float bv = bias ? bias[col] : 0.0f;
#pragma unroll
    for (int i = 0; i < 2; ++i) {
#pragma unroll
      for (int r = 0; r < 8; ++r) {
        int row = m0 + i * 16 + r + 8 * half;                    // C layout
        float v = acc[i][j][r] + bv;
        if (relu) v = fmaxf(v, 0.0f);
        size_t idx = (size_t)row * N + col;
        if (Cf) Cf[idx] = v;
        if (Cb) Cb[idx] = to_bf16(v);
      }
    }
  }
}

// ---------------------------------------------------------------------------
// Head packs: [M][D] fp32 -> per-(b,h) bf16 tensors.
// pack_heads: dst[b][h][t][dd]  (Q as A-matrix rows; K natural = Kt for B-frag)
// pack_vt:    dst[b][h][dd][t]  (V^T as [N][K] for P*V B-frag)
// ---------------------------------------------------------------------------
__global__ void pack_heads_kernel(const float* __restrict__ src, bf16* __restrict__ dst) {
  size_t id = (size_t)blockIdx.x * blockDim.x + threadIdx.x;   // < MROWS*DMODEL
  int dd = (int)(id & 63);
  int t  = (int)((id >> 6) & (SEQ - 1));
  int bh = (int)(id >> 16);
  int b = bh >> 4, hh = bh & 15;
  dst[id] = to_bf16(src[((size_t)(b * SEQ + t)) * DMODEL + hh * HDIM + dd]);
}

__global__ void pack_vt_kernel(const float* __restrict__ src, bf16* __restrict__ dst) {
  size_t id = (size_t)blockIdx.x * blockDim.x + threadIdx.x;   // < MROWS*DMODEL
  int t  = (int)(id & (SEQ - 1));
  int dd = (int)((id >> 10) & 63);
  int bh = (int)(id >> 16);
  int b = bh >> 4, hh = bh & 15;
  dst[id] = to_bf16(src[((size_t)(b * SEQ + t)) * DMODEL + hh * HDIM + dd]);
}

// ---------------------------------------------------------------------------
// Flash attention, one wave per (b, h, 16-query tile).
// S = Q K^T * 0.125 + (+1.0 allowed / -inf masked)   [reference's tril mask]
// Online softmax; P staged via LDS to convert C-layout -> A-layout; O = P V.
// ---------------------------------------------------------------------------
__global__ void attn_kernel(const bf16* __restrict__ Q, const bf16* __restrict__ Kh,
                            const bf16* __restrict__ Vt, float* __restrict__ O) {
  int lane = threadIdx.x;
  int half = lane >> 4, ln = lane & 15;
  int q0 = blockIdx.x * 16;
  int bh = blockIdx.z * NHEAD + blockIdx.y;
  const bf16* q = Q  + (size_t)bh * SEQ * HDIM;   // [T][64]
  const bf16* k = Kh + (size_t)bh * SEQ * HDIM;   // [T][64] == (K^T)t layout
  const bf16* v = Vt + (size_t)bh * HDIM * SEQ;   // [64][T]

  v16bf qa[2];
#pragma unroll
  for (int ks = 0; ks < 2; ++ks) {
    const bf16* qp = q + (size_t)(q0 + ln) * HDIM + ks * 32 + 16 * half;
    qa[ks] = make_a(*(const v8bf*)qp, *(const v8bf*)(qp + 8));
  }

  v8f o[4] = {};
  float mr[8], lr[8];
#pragma unroll
  for (int r = 0; r < 8; ++r) { mr[r] = -1e30f; lr[r] = 0.0f; }

  __shared__ __align__(32) bf16 plds[16 * 32];

  int kbmax = (q0 + 15) >> 5;
  for (int kb = 0; kb <= kbmax; ++kb) {
    int key0 = kb * 32;
    v8f s0 = {}, s1 = {};
#pragma unroll
    for (int ks = 0; ks < 2; ++ks) {            // contraction over hd=64
      v16bf b0 = *(const v16bf*)(k + (size_t)(key0 + ln) * HDIM + ks * 32 + 16 * half);
      v16bf b1 = *(const v16bf*)(k + (size_t)(key0 + 16 + ln) * HDIM + ks * 32 + 16 * half);
      s0 = __builtin_amdgcn_wmma_f32_16x16x32_bf16(false, qa[ks], false, b0, (short)0, s0, false, false);
      s1 = __builtin_amdgcn_wmma_f32_16x16x32_bf16(false, qa[ks], false, b1, (short)0, s1, false, false);
    }

    float p0[8], p1[8], mnew[8], alpha[8];
#pragma unroll
    for (int r = 0; r < 8; ++r) {               // scale + causal mask (+1.0)
      int qrow = q0 + r + 8 * half;
      int c0 = key0 + ln, c1 = key0 + 16 + ln;
      p0[r] = (c0 <= qrow) ? s0[r] * 0.125f + 1.0f : -1e30f;
      p1[r] = (c1 <= qrow) ? s1[r] * 0.125f + 1.0f : -1e30f;
    }
#pragma unroll
    for (int r = 0; r < 8; ++r) {               // row max (16 lanes / half)
      float t = fmaxf(p0[r], p1[r]);
      t = fmaxf(t, __shfl_xor(t, 1, 32));
      t = fmaxf(t, __shfl_xor(t, 2, 32));
      t = fmaxf(t, __shfl_xor(t, 4, 32));
      t = fmaxf(t, __shfl_xor(t, 8, 32));
      mnew[r] = fmaxf(mr[r], t);
      alpha[r] = __expf(mr[r] - mnew[r]);
      mr[r] = mnew[r];
    }
#pragma unroll
    for (int r = 0; r < 8; ++r) {               // exp + row sum
      p0[r] = __expf(p0[r] - mnew[r]);
      p1[r] = __expf(p1[r] - mnew[r]);
      float t = p0[r] + p1[r];
      t += __shfl_xor(t, 1, 32);
      t += __shfl_xor(t, 2, 32);
      t += __shfl_xor(t, 4, 32);
      t += __shfl_xor(t, 8, 32);
      lr[r] = lr[r] * alpha[r] + t;
    }
#pragma unroll
    for (int n = 0; n < 4; ++n)
#pragma unroll
      for (int r = 0; r < 8; ++r) o[n][r] *= alpha[r];

#pragma unroll
    for (int r = 0; r < 8; ++r) {               // stage P (C-layout -> LDS)
      plds[(r + 8 * half) * 32 + ln]      = to_bf16(p0[r]);
      plds[(r + 8 * half) * 32 + 16 + ln] = to_bf16(p1[r]);
    }
    __syncthreads();
    const bf16* pp = plds + ln * 32 + 16 * half; // read back in A-layout
    v16bf pa = make_a(*(const v8bf*)pp, *(const v8bf*)(pp + 8));
    __syncthreads();

#pragma unroll
    for (int n = 0; n < 4; ++n) {               // O += P * V  (N = hd)
      v16bf vb = *(const v16bf*)(v + (size_t)(n * 16 + ln) * SEQ + key0 + 16 * half);
      o[n] = __builtin_amdgcn_wmma_f32_16x16x32_bf16(false, pa, false, vb, (short)0, o[n], false, false);
    }
  }

  float* obase = O + (size_t)bh * SEQ * HDIM;
#pragma unroll
  for (int r = 0; r < 8; ++r) {
    float inv = 1.0f / lr[r];
#pragma unroll
    for (int n = 0; n < 4; ++n)
      obase[(size_t)(q0 + r + 8 * half) * HDIM + n * 16 + ln] = o[n][r] * inv;
  }
}

// ---------------------------------------------------------------------------
// LayerNorm over D=1024: y = (x+res - mean)/sqrt(var+eps)*gamma + beta.
// attn_gather==1: x read from [b][h][t][hd] attention-output layout.
// Emits fp32 + bf16 outputs (bf16 feeds next GEMM directly).
// ---------------------------------------------------------------------------
__global__ void ln_kernel(const float* __restrict__ xin, const float* __restrict__ res,
                          const float* __restrict__ gamma, const float* __restrict__ beta,
                          float* __restrict__ yf, bf16* __restrict__ yb, int attn_gather) {
  __shared__ float sh1[256], sh2[256];
  int row = blockIdx.x, tid = threadIdx.x;
  float vals[4], s = 0.0f, s2 = 0.0f;
#pragma unroll
  for (int j = 0; j < 4; ++j) {
    int d = tid + j * 256;
    float xv;
    if (attn_gather) {
      int b = row >> 10, t = row & 1023, head = d >> 6, off = d & 63;
      xv = xin[(((size_t)(b * NHEAD + head)) * SEQ + t) * HDIM + off];
    } else {
      xv = xin[(size_t)row * DMODEL + d];
    }
    xv += res[(size_t)row * DMODEL + d];
    vals[j] = xv; s += xv; s2 += xv * xv;
  }
  sh1[tid] = s; sh2[tid] = s2;
  __syncthreads();
  for (int off = 128; off > 0; off >>= 1) {
    if (tid < off) { sh1[tid] += sh1[tid + off]; sh2[tid] += sh2[tid + off]; }
    __syncthreads();
  }
  float mean = sh1[0] * (1.0f / DMODEL);
  float var  = sh2[0] * (1.0f / DMODEL) - mean * mean;
  float rst  = rsqrtf(var + 1e-5f);
#pragma unroll
  for (int j = 0; j < 4; ++j) {
    int d = tid + j * 256;
    float y = (vals[j] - mean) * rst * gamma[d] + beta[d];
    yf[(size_t)row * DMODEL + d] = y;
    yb[(size_t)row * DMODEL + d] = to_bf16(y);
  }
}

// ---------------------------------------------------------------------------
// Host orchestration
// ---------------------------------------------------------------------------
extern "C" void kernel_launch(void* const* d_in, const int* in_sizes, int n_in,
                              void* d_out, int out_size, void* d_ws, size_t ws_size,
                              hipStream_t stream) {
  (void)in_sizes; (void)n_in; (void)out_size; (void)ws_size;
  const int*   x    = (const int*)  d_in[0];
  const float* tok  = (const float*)d_in[1];
  const float* pos  = (const float*)d_in[2];
  const float* Wq   = (const float*)d_in[3];
  const float* bq   = (const float*)d_in[4];
  const float* Wk   = (const float*)d_in[5];
  const float* bk   = (const float*)d_in[6];
  const float* Wv   = (const float*)d_in[7];
  const float* bv   = (const float*)d_in[8];
  const float* ln1s = (const float*)d_in[9];
  const float* ln1b = (const float*)d_in[10];
  const float* W1   = (const float*)d_in[11];
  const float* b1   = (const float*)d_in[12];
  const float* W2   = (const float*)d_in[13];
  const float* b2   = (const float*)d_in[14];
  const float* ln2s = (const float*)d_in[15];
  const float* ln2b = (const float*)d_in[16];
  const float* Wout = (const float*)d_in[17];
  const float* bout = (const float*)d_in[18];

  const int M = MROWS, D = DMODEL, V = VOCAB;

  char* p = (char*)d_ws;
  auto alloc = [&](size_t bytes) -> void* {
    void* r = (void*)p;
    p += (bytes + 255) & ~(size_t)255;
    return r;
  };
  float* h   = (float*)alloc((size_t)M * D * 4);       // residual
  bf16*  hb  = (bf16*) alloc((size_t)M * D * 2);
  float* hi  = (float*)alloc((size_t)M * D * 4);       // post-ln1
  bf16*  hib = (bf16*) alloc((size_t)M * D * 2);
  float* qf  = (float*)alloc((size_t)M * D * 4);
  float* kf  = (float*)alloc((size_t)M * D * 4);
  float* vf  = (float*)alloc((size_t)M * D * 4);
  bf16*  qh  = (bf16*) alloc((size_t)M * D * 2);       // [b][h][t][64]
  bf16*  kh  = (bf16*) alloc((size_t)M * D * 2);       // [b][h][t][64]
  bf16*  vt  = (bf16*) alloc((size_t)M * D * 2);       // [b][h][64][t]
  float* ao  = (float*)alloc((size_t)M * D * 4);       // attention out
  bf16*  mid = (bf16*) alloc((size_t)M * 4 * D * 2);   // MLP mid (relu'd)
  float* ff  = (float*)alloc((size_t)M * D * 4);       // MLP out
  bf16*  wT  = (bf16*) alloc((size_t)4 * D * D * 2);   // transposed weight scratch
  bf16*  woT = (bf16*) alloc((size_t)D * V * 2);       // transposed Wout

  auto packW = [&](const float* W, bf16* dst, int K, int N) {
    pack_wt_kernel<<<dim3((unsigned)(K / 32), (unsigned)(N / 32)), 256, 0, stream>>>(W, dst, K, N);
  };
  auto gemm = [&](const bf16* A, const bf16* Bt, const float* bias,
                  float* Cf, bf16* Cb, int Mm, int Nn, int Kk, int relu) {
    dim3 g((unsigned)(Mm / 32), (unsigned)(Nn / 256));
    gemm_bf16_wmma<<<g, 128, 0, stream>>>(A, Bt, bias, Cf, Cb, Mm, Nn, Kk, relu);
  };

  embed_kernel<<<M, 256, 0, stream>>>(x, tok, pos, h, hb);

  const int totMD = M * D;  // 2,097,152
  for (int l = 0; l < NLAYER; ++l) {
    // QKV projections (post-LN arch: read raw residual)
    packW(Wq + (size_t)l * D * D, wT, D, D);
    gemm(hb, wT, bq + (size_t)l * D, qf, nullptr, M, D, D, 0);
    packW(Wk + (size_t)l * D * D, wT, D, D);
    gemm(hb, wT, bk + (size_t)l * D, kf, nullptr, M, D, D, 0);
    packW(Wv + (size_t)l * D * D, wT, D, D);
    gemm(hb, wT, bv + (size_t)l * D, vf, nullptr, M, D, D, 0);

    pack_heads_kernel<<<totMD / 256, 256, 0, stream>>>(qf, qh);
    pack_heads_kernel<<<totMD / 256, 256, 0, stream>>>(kf, kh);
    pack_vt_kernel  <<<totMD / 256, 256, 0, stream>>>(vf, vt);

    attn_kernel<<<dim3(SEQ / 16, NHEAD, BATCH), 32, 0, stream>>>(qh, kh, vt, ao);

    // hi = LN1(attn_out + h)
    ln_kernel<<<M, 256, 0, stream>>>(ao, h, ln1s + (size_t)l * D, ln1b + (size_t)l * D,
                                     hi, hib, 1);
    // MLP
    packW(W1 + (size_t)l * D * 4 * D, wT, D, 4 * D);
    gemm(hib, wT, b1 + (size_t)l * 4 * D, nullptr, mid, M, 4 * D, D, 1);
    packW(W2 + (size_t)l * 4 * D * D, wT, 4 * D, D);
    gemm(mid, wT, b2 + (size_t)l * D, ff, nullptr, M, D, 4 * D, 0);
    // h = LN2(mlp_out + hi)
    ln_kernel<<<M, 256, 0, stream>>>(ff, hi, ln2s + (size_t)l * D, ln2b + (size_t)l * D,
                                     h, hb, 0);
  }

  // LM head
  packW(Wout, woT, D, V);
  gemm(hb, woT, bout, (float*)d_out, nullptr, M, V, D, 0);
}
